// hRQA_14156212208271
// MI455X (gfx1250) — compile-verified, matrix operand order checked by reading, not run on hardware
//
#include <hip/hip_runtime.h>

// ---------------------------------------------------------------------------
// Problem constants
// ---------------------------------------------------------------------------
#define NROWS 8192
#define DIM   256
#define INV_SIGMA 3.33333333333f          // 1/0.3
#define LOGC      0.28503428291f          // -log(0.3) - 0.5*log(2*pi)

// GEMM tiling
#define ROWS_PER_BLOCK  64                // 4 waves x 16-row WMMA tiles
#define WAVES_PER_BLOCK 4
#define THREADS         (WAVES_PER_BLOCK * 32)
#define NCOLTILES       (NROWS / 16)      // 512
#define NSLICES         8                 // column slices per row-block
#define TILES_PER_SLICE (NCOLTILES / NSLICES) // 64
#define LDS_STRIDE      264               // 256 + 8 pad (bf16 elems): 132 dwords,
                                          // 132 % 64 == 4 -> lane l hits banks 4l..4l+3

typedef __attribute__((ext_vector_type(16))) __bf16 v16bf;
typedef __attribute__((ext_vector_type(8)))  float  v8f;

// ---------------------------------------------------------------------------
// bf16 helpers (bit-level RNE; avoids scalar __bf16 conversions)
// ---------------------------------------------------------------------------
__device__ __forceinline__ unsigned short f32_to_bf16_rne(float f) {
    unsigned u = __float_as_uint(f);
    u += 0x7FFFu + ((u >> 16) & 1u);
    return (unsigned short)(u >> 16);
}
__device__ __forceinline__ float bf16_to_f32(unsigned short h) {
    return __uint_as_float(((unsigned)h) << 16);
}

// order-preserving float <-> uint key for unsigned atomicMin
__device__ __forceinline__ unsigned f2key(float f) {
    unsigned b = __float_as_uint(f);
    return (b & 0x80000000u) ? ~b : (b | 0x80000000u);
}
__device__ __forceinline__ float key2f(unsigned u) {
    unsigned b = (u & 0x80000000u) ? (u ^ 0x80000000u) : ~u;
    return __uint_as_float(b);
}

// ---------------------------------------------------------------------------
// Kernel 0: init min-keys to "+inf" key
// ---------------------------------------------------------------------------
__global__ void init_keys_kernel(unsigned* __restrict__ keys) {
    keys[blockIdx.x * blockDim.x + threadIdx.x] = 0xFFFFFFFFu;
}

// ---------------------------------------------------------------------------
// Kernel 1: row-normalize Ex/Ey, split normalized value into bf16 hi + lo.
// One wave32 per row; lane reduction via shuffles.
// ---------------------------------------------------------------------------
__global__ void normalize_split_kernel(const float* __restrict__ Ex,
                                       const float* __restrict__ Ey,
                                       unsigned short* __restrict__ Xhi,
                                       unsigned short* __restrict__ Xlo,
                                       unsigned short* __restrict__ Yhi,
                                       unsigned short* __restrict__ Ylo) {
    const int wave = (blockIdx.x * blockDim.x + threadIdx.x) >> 5;
    const int lane = threadIdx.x & 31;

    const float* src;
    unsigned short *hi, *lo;
    int row;
    if (wave < NROWS) { src = Ex; hi = Xhi; lo = Xlo; row = wave; }
    else              { src = Ey; hi = Yhi; lo = Ylo; row = wave - NROWS; }

    const float* p = src + (size_t)row * DIM;
    float v[DIM / 32];
    float ss = 0.0f;
#pragma unroll
    for (int i = 0; i < DIM / 32; ++i) {
        v[i] = p[lane + 32 * i];
        ss = fmaf(v[i], v[i], ss);
    }
#pragma unroll
    for (int off = 16; off > 0; off >>= 1)
        ss += __shfl_xor(ss, off, 32);

    const float inv = 1.0f / fmaxf(sqrtf(ss), 1e-8f);

#pragma unroll
    for (int i = 0; i < DIM / 32; ++i) {
        const float x = v[i] * inv;
        const unsigned short h = f32_to_bf16_rne(x);
        const unsigned short l = f32_to_bf16_rne(x - bf16_to_f32(h));
        const size_t idx = (size_t)row * DIM + lane + 32 * i;
        hi[idx] = h;
        lo[idx] = l;
    }
}

// ---------------------------------------------------------------------------
// Fragment gather per ISA 7.12.2 (16-bit A/B 16x32 layout):
//   lanes 0-15 : K = k0+0..7   then k0+16..23
//   lanes 16-31: K = k0+8..15  then k0+24..31
// Two ds_load_b128 per fragment from the padded LDS tile.
// ---------------------------------------------------------------------------
__device__ __forceinline__ v16bf lds_frag(const unsigned short* p) {
    union { uint4 q[2]; v16bf v; } f;
    f.q[0] = *(const uint4*)(p);
    f.q[1] = *(const uint4*)(p + 16);
    return f.v;
}

// ---------------------------------------------------------------------------
// Kernel 2: fused split-bf16 WMMA GEMM + Normal(1,sigma) log-pdf + row-min.
//   grid = (NROWS/64 row-blocks, 8 column slices), block = 4 waves.
//   A tile (64 rows, hi+lo) LDS-resident; B tile (16 cols, hi+lo) staged
//   through LDS with a register double-buffer so global latency overlaps
//   the 24-WMMA compute block.
// ---------------------------------------------------------------------------
__global__ __launch_bounds__(THREADS) void cosmin_wmma_kernel(
    const unsigned short* __restrict__ Xhi,
    const unsigned short* __restrict__ Xlo,
    const unsigned short* __restrict__ Yhi,
    const unsigned short* __restrict__ Ylo,
    unsigned* __restrict__ keys) {
    __shared__ unsigned short lAh[ROWS_PER_BLOCK * LDS_STRIDE];
    __shared__ unsigned short lAl[ROWS_PER_BLOCK * LDS_STRIDE];
    __shared__ unsigned short lBh[16 * LDS_STRIDE];
    __shared__ unsigned short lBl[16 * LDS_STRIDE];

    const int tid      = threadIdx.x;
    const int lane     = tid & 31;
    const int wave     = tid >> 5;                     // 0..3
    const int rowBase  = blockIdx.x * ROWS_PER_BLOCK;  // 64-row block
    const int tileBase = blockIdx.y * TILES_PER_SLICE; // first column tile

    // ---- cooperative A tile -> LDS (64 rows, hi + lo) ----
#pragma unroll 2
    for (int idx = tid; idx < ROWS_PER_BLOCK * (DIM / 8); idx += THREADS) {
        const int row = idx >> 5;                      // 32 16B-chunks per row
        const int ch  = idx & 31;
        const size_t s = (size_t)(rowBase + row) * DIM + ch * 8;
        *(uint4*)&lAh[row * LDS_STRIDE + ch * 8] = *(const uint4*)&Xhi[s];
        *(uint4*)&lAl[row * LDS_STRIDE + ch * 8] = *(const uint4*)&Xlo[s];
    }

    // ---- B tile register double-buffer (16 rows x 256 elems, hi + lo) ----
    uint4 rh[4], rl[4];
    auto prefetchB = [&](int t) {
        const int colBase = (tileBase + t) * 16;
#pragma unroll
        for (int i = 0; i < 4; ++i) {
            const int cid = tid + THREADS * i;         // 0..511
            const int row = cid >> 5;
            const int ch  = cid & 31;
            const size_t s = (size_t)(colBase + row) * DIM + ch * 8;
            rh[i] = *(const uint4*)&Yhi[s];
            rl[i] = *(const uint4*)&Ylo[s];
        }
    };
    prefetchB(0);

    float minv[8];
#pragma unroll
    for (int r = 0; r < 8; ++r) minv[r] = 3.402823466e38f;

    const int hiSel = (lane >> 4) << 3;                // 0 or 8
    const int aOff  = (wave * 16 + (lane & 15)) * LDS_STRIDE + hiSel;
    const int bOff  = (lane & 15) * LDS_STRIDE + hiSel;

    for (int t = 0; t < TILES_PER_SLICE; ++t) {
        __syncthreads();                               // LDS B free to overwrite
#pragma unroll
        for (int i = 0; i < 4; ++i) {
            const int cid = tid + THREADS * i;
            const int row = cid >> 5;
            const int ch  = cid & 31;
            *(uint4*)&lBh[row * LDS_STRIDE + ch * 8] = rh[i];
            *(uint4*)&lBl[row * LDS_STRIDE + ch * 8] = rl[i];
        }
        __syncthreads();
        if (t + 1 < TILES_PER_SLICE) prefetchB(t + 1); // overlaps compute below

        v8f acc = {};
#pragma unroll
        for (int c = 0; c < DIM / 32; ++c) {
            const int k0 = c * 32;
            v16bf ah = lds_frag(&lAh[aOff + k0]);
            v16bf al = lds_frag(&lAl[aOff + k0]);
            v16bf bh = lds_frag(&lBh[bOff + k0]);
            v16bf bl = lds_frag(&lBl[bOff + k0]);
            // acc += Al*Bh + Ah*Bl + Ah*Bh  (split-bf16 ~f32 dot product)
            acc = __builtin_amdgcn_wmma_f32_16x16x32_bf16(
                      false, al, false, bh, (short)0, acc, false, false);
            acc = __builtin_amdgcn_wmma_f32_16x16x32_bf16(
                      false, ah, false, bl, (short)0, acc, false, false);
            acc = __builtin_amdgcn_wmma_f32_16x16x32_bf16(
                      false, ah, false, bh, (short)0, acc, false, false);
        }

        // epilogue: Normal(1, 0.3) log_prob -> C = -exp(logp)*logp -> running min
#pragma unroll
        for (int r = 0; r < 8; ++r) {
            const float m    = acc[r];
            const float z    = (m - 1.0f) * INV_SIGMA;
            const float logp = fmaf(-0.5f * z, z, LOGC);
            const float C    = -__expf(logp) * logp;
            minv[r] = fminf(minv[r], C);
        }
    }

    // reduce the min across the 16 lanes that share each output row
#pragma unroll
    for (int r = 0; r < 8; ++r) {
#pragma unroll
        for (int off = 1; off < 16; off <<= 1)
            minv[r] = fminf(minv[r], __shfl_xor(minv[r], off, 32));
    }

    // lane 0 holds rows 0..7 of the wave tile (VGPR r), lane 16 rows 8..15
    if ((lane & 15) == 0) {
        const int base = rowBase + wave * 16 + ((lane >> 4) << 3);
#pragma unroll
        for (int r = 0; r < 8; ++r)
            atomicMin(&keys[base + r], f2key(minv[r]));
    }
}

// ---------------------------------------------------------------------------
// Kernel 3: uint key -> float output
// ---------------------------------------------------------------------------
__global__ void finalize_kernel(const unsigned* __restrict__ keys,
                                float* __restrict__ out) {
    const int i = blockIdx.x * blockDim.x + threadIdx.x;
    out[i] = key2f(keys[i]);
}

// ---------------------------------------------------------------------------
// Host entry
// ---------------------------------------------------------------------------
extern "C" void kernel_launch(void* const* d_in, const int* in_sizes, int n_in,
                              void* d_out, int out_size, void* d_ws, size_t ws_size,
                              hipStream_t stream) {
    const float* Ex = (const float*)d_in[0];
    const float* Ey = (const float*)d_in[1];

    const size_t elems = (size_t)NROWS * DIM;
    unsigned short* Xhi = (unsigned short*)d_ws;
    unsigned short* Xlo = Xhi + elems;
    unsigned short* Yhi = Xlo + elems;
    unsigned short* Ylo = Yhi + elems;
    unsigned*      keys = (unsigned*)(Ylo + elems);

    // 0) init per-row min keys
    hipLaunchKernelGGL(init_keys_kernel, dim3(NROWS / 256), dim3(256), 0, stream,
                       keys);
    // 1) normalize + split to bf16 hi/lo: 2*8192 rows, 1 wave each
    hipLaunchKernelGGL(normalize_split_kernel, dim3((2 * NROWS) / 8), dim3(256), 0,
                       stream, Ex, Ey, Xhi, Xlo, Yhi, Ylo);
    // 2) fused WMMA GEMM + logpdf + row-min
    hipLaunchKernelGGL(cosmin_wmma_kernel,
                       dim3(NROWS / ROWS_PER_BLOCK, NSLICES), dim3(THREADS), 0,
                       stream, Xhi, Xlo, Yhi, Ylo, keys);
    // 3) keys -> float out
    hipLaunchKernelGGL(finalize_kernel, dim3(NROWS / 256), dim3(256), 0, stream,
                       keys, (float*)d_out);
}
